// MultiHeadQGFDLayer_90323162235322
// MI455X (gfx1250) — matrix operand order repaired and checked by reading, not compiled, not run on hardware
//
#include <hip/hip_runtime.h>
#include <cstddef>
#include <cstdint>

typedef __attribute__((ext_vector_type(16))) __bf16 v16bf;
typedef __attribute__((ext_vector_type(8)))  float  v8f;

union AB16 { uint4 q[2]; v16bf v; };

constexpr int Bn  = 8;
constexpr int Ln  = 512;
constexpr int Dn  = 1024;
constexpr int Hn  = 16;
constexpr int HDn = 64;
constexpr int BHn = Bn * Hn;                 // 128
constexpr int MLD = Bn * Ln;                 // 4096 rows for projections
constexpr size_t NT = (size_t)Bn * Ln * Dn;  // 4,194,304 elements
constexpr size_t WSZ = (size_t)Dn * Dn;      // 1,048,576
constexpr size_t PP  = (size_t)BHn * Ln * Ln; // 33,554,432 per-head matrix elems

__device__ __forceinline__ unsigned short f2bf(float x) {
  unsigned u = __float_as_uint(x);
  u += 0x7FFFu + ((u >> 16) & 1u);           // round-to-nearest-even
  return (unsigned short)(u >> 16);
}
__device__ __forceinline__ float bf2f(unsigned short h) {
  return __uint_as_float(((unsigned)h) << 16);
}

// ---------------- utility kernels ----------------

__global__ __launch_bounds__(256) void f2bf_kernel(const float* __restrict__ in,
                                                   unsigned short* __restrict__ out, int n) {
  int i = blockIdx.x * 256 + threadIdx.x;
  if (i < n) out[i] = f2bf(in[i]);
}

__global__ void init_flags(unsigned int* fl) {
  if (threadIdx.x < 8) fl[threadIdx.x] = 0u;
}

__global__ void update_done(unsigned int* fl, int i) {
  float d = __uint_as_float(fl[i]);
  if (d < 1e-5f) fl[4] = 1u;                 // done |= cond_i
}

// normalize each (b,l,h) 64-vector of K (bf16 in, bf16 out); one wave per vector
__global__ __launch_bounds__(256) void knorm_kernel(const unsigned short* __restrict__ Kb,
                                                    unsigned short* __restrict__ Knb) {
  const int lane = threadIdx.x & 31;
  const int id = blockIdx.x * 8 + (threadIdx.x >> 5);   // 0..65535
  const int h = id & 15, l = (id >> 4) & 511, b = id >> 13;
  const size_t base = (size_t)b * Ln * Dn + (size_t)l * Dn + (size_t)h * HDn;
  float x0 = bf2f(Kb[base + 2 * lane]);
  float x1 = bf2f(Kb[base + 2 * lane + 1]);
  float s = x0 * x0 + x1 * x1;
  for (int o = 16; o; o >>= 1) s += __shfl_xor(s, o, 32);
  float inv = 1.0f / fmaxf(sqrtf(s), 1e-6f);
  Knb[base + 2 * lane]     = f2bf(x0 * inv);
  Knb[base + 2 * lane + 1] = f2bf(x1 * inv);
}

// row softmax over 512 cols; one wave per row. Optional jitter (for P).
__global__ __launch_bounds__(256) void softmax_kernel(const float* __restrict__ raw,
                                                      float* __restrict__ outf,
                                                      unsigned short* __restrict__ outbf,
                                                      int jitter) {
  const int lane = threadIdx.x & 31;
  const long long row = (long long)blockIdx.x * 8 + (threadIdx.x >> 5);
  const float* r = raw + row * Ln;
  float x[16];
  float m = -3.4e38f;
  for (int j = 0; j < 16; ++j) { x[j] = r[lane + 32 * j]; m = fmaxf(m, x[j]); }
  for (int o = 16; o; o >>= 1) m = fmaxf(m, __shfl_xor(m, o, 32));
  float s = 0.f;
  for (int j = 0; j < 16; ++j) { x[j] = __expf(x[j] - m); s += x[j]; }
  for (int o = 16; o; o >>= 1) s += __shfl_xor(s, o, 32);
  float inv = 1.0f / s;
  for (int j = 0; j < 16; ++j) {
    float p = x[j] * inv;
    if (jitter) p = p * (1.0f - 1e-6f) + (1e-6f / 512.0f);
    long long idx = row * Ln + lane + 32 * j;
    if (outf)  outf[idx]  = p;
    if (outbf) outbf[idx] = f2bf(p);
  }
}

// ---------------- unified WMMA GEMM (double-buffered LDS) ----------------
// Block tile 128 x BN, BK = 32, 256 threads = 8 waves (4x2), wave tile 32 x BN/2.
// MODE 0: C_bf16 = A*B + bias           (projections;  A bf16)
// MODE 1: C_f32  = A*B + bias           (output proj;  A bf16)
// MODE 2: diffusion: out = 0.98*p0 + 0.02*(p@P); fp32 out; |diff| atomicMax; done-copy path
// MODE 3: attn = p@V -> bf16 into (B,L,D) layout   (A fp32)
// MODE 4: batched scaled scores: out_f32 = scale * A*B^T   (A,B bf16, BTRANS)
template<int BN, int MODE, bool AF32, bool BTRANS>
__global__ __launch_bounds__(256)
__attribute__((amdgpu_waves_per_eu(2, 8)))
void gemm_kernel(
    const void* __restrict__ Aptr, const unsigned short* __restrict__ Bptr,
    const float* __restrict__ bias, const float* __restrict__ p0f,
    const float* __restrict__ prevf, float* __restrict__ outf,
    unsigned short* __restrict__ outbf, unsigned int* __restrict__ diffslot,
    const unsigned int* __restrict__ doneflag,
    int M, int N, int K, int lda, int ldb, float scale) {
  constexpr int BM = 128, BK = 32;
  constexpr int LSA = BK + 8, LSB = BK + 8;
  const int tid = threadIdx.x, lane = tid & 31, wid = tid >> 5;
  const int bh = blockIdx.z;
  const int m0 = blockIdx.y * BM, n0 = blockIdx.x * BN;

  size_t aoff = 0, boff = 0;
  if constexpr (MODE == 2) {
    aoff = (size_t)bh * (Ln * Ln); boff = aoff;
  } else if constexpr (MODE == 3) {
    aoff = (size_t)bh * (Ln * Ln);
    boff = (size_t)(bh >> 4) * (Ln * Dn) + (size_t)(bh & 15) * HDn;
  } else if constexpr (MODE == 4) {
    aoff = (size_t)(bh >> 4) * (Ln * Dn) + (size_t)(bh & 15) * HDn;
    boff = aoff;
  }

  if constexpr (MODE == 2) {
    const float* Af = (const float*)Aptr;
    if (*doneflag) {                          // p frozen: pass current p through
      for (int i = tid; i < BM * BN; i += 256) {
        int r = i / BN, c = i % BN;
        size_t idx = aoff + (size_t)(m0 + r) * N + (n0 + c);
        outf[idx] = Af[idx];
      }
      return;
    }
  }

  __shared__ unsigned short lds_a[2][BM * LSA];
  __shared__ unsigned short lds_b[2][BN * LSB];

  constexpr int WN = BN / 2, NF = WN / 16, MF = 2;
  const int wrow = wid & 3, wcol = wid >> 2;
  const int wm0 = wrow * 32, wn0 = wcol * WN;
  const int half = lane >> 4, l16 = lane & 15;

  const float*          Agf = (const float*)Aptr + aoff;          // valid iff AF32
  const unsigned short* Agb = (const unsigned short*)Aptr + aoff; // valid iff !AF32
  const unsigned short* Bg  = Bptr + boff;

  float4 fa[4];      // A staging (fp32 path)
  uint4  ua4[2];     // A staging (bf16 path)
  uint4  ub4[BN / 64];

  auto gload = [&](int kb) {
    if constexpr (AF32) {
#pragma unroll
      for (int i = 0; i < 4; ++i) {
        int e = tid + i * 256;                 // 1024 float4 chunks
        int row = e >> 3, ch = e & 7;
        fa[i] = *(const float4*)(Agf + (size_t)(m0 + row) * lda + kb * BK + ch * 4);
      }
    } else {
#pragma unroll
      for (int i = 0; i < 2; ++i) {
        int e = tid + i * 256;                 // 512 uint4 chunks
        int row = e >> 2, ch = e & 3;
        ua4[i] = *(const uint4*)(Agb + (size_t)(m0 + row) * lda + kb * BK + ch * 8);
      }
    }
#pragma unroll
    for (int i = 0; i < BN / 64; ++i) {
      int e = tid + i * 256;
      if constexpr (BTRANS) {                  // B element (k,n) lives at Bg[n*ldb + k]
        int nl = e >> 2, kch = e & 3;
        ub4[i] = *(const uint4*)(Bg + (size_t)(n0 + nl) * ldb + kb * BK + kch * 8);
      } else {
        int kr = e / (BN / 8), ch = e % (BN / 8);
        ub4[i] = *(const uint4*)(Bg + (size_t)(kb * BK + kr) * ldb + n0 + ch * 8);
      }
    }
  };

  auto lstore = [&](int buf) {
    if constexpr (AF32) {
#pragma unroll
      for (int i = 0; i < 4; ++i) {
        int e = tid + i * 256;
        int row = e >> 3, ch = e & 7;
        uint2 pk;
        pk.x = (unsigned)f2bf(fa[i].x) | ((unsigned)f2bf(fa[i].y) << 16);
        pk.y = (unsigned)f2bf(fa[i].z) | ((unsigned)f2bf(fa[i].w) << 16);
        *(uint2*)&lds_a[buf][row * LSA + ch * 4] = pk;
      }
    } else {
#pragma unroll
      for (int i = 0; i < 2; ++i) {
        int e = tid + i * 256;
        int row = e >> 2, ch = e & 3;
        *(uint4*)&lds_a[buf][row * LSA + ch * 8] = ua4[i];
      }
    }
#pragma unroll
    for (int i = 0; i < BN / 64; ++i) {
      int e = tid + i * 256;
      if constexpr (BTRANS) {
        int nl = e >> 2, kch = e & 3;
        *(uint4*)&lds_b[buf][nl * LSB + kch * 8] = ub4[i];
      } else {                                 // scatter-transpose into lds_b[n][k]
        int kr = e / (BN / 8), ch = e % (BN / 8);
        int nl = ch * 8;
        uint4 q = ub4[i];
        lds_b[buf][(nl + 0) * LSB + kr] = (unsigned short)(q.x);
        lds_b[buf][(nl + 1) * LSB + kr] = (unsigned short)(q.x >> 16);
        lds_b[buf][(nl + 2) * LSB + kr] = (unsigned short)(q.y);
        lds_b[buf][(nl + 3) * LSB + kr] = (unsigned short)(q.y >> 16);
        lds_b[buf][(nl + 4) * LSB + kr] = (unsigned short)(q.z);
        lds_b[buf][(nl + 5) * LSB + kr] = (unsigned short)(q.z >> 16);
        lds_b[buf][(nl + 6) * LSB + kr] = (unsigned short)(q.w);
        lds_b[buf][(nl + 7) * LSB + kr] = (unsigned short)(q.w >> 16);
      }
    }
  };

  v8f acc[MF][NF];
  for (int i = 0; i < MF; ++i)
    for (int j = 0; j < NF; ++j)
      for (int e = 0; e < 8; ++e) acc[i][j][e] = 0.0f;

  const int nkb = K / BK;

  // prologue: stage tile 0
  gload(0);
  lstore(0);
  __syncthreads();

  int cur = 0;
  for (int kb = 0; kb < nkb; ++kb) {
    const bool more = (kb + 1) < nkb;
    if (more) {
      gload(kb + 1);                           // global loads in flight during WMMAs
      if (kb + 2 < nkb) {                      // pull K+2 tiles toward L2 (global_prefetch)
        if constexpr (AF32)
          __builtin_prefetch((const void*)(Agf + (size_t)(m0 + (tid >> 3)) * lda + (kb + 2) * BK), 0, 0);
        else
          __builtin_prefetch((const void*)(Agb + (size_t)(m0 + (tid >> 2)) * lda + (kb + 2) * BK), 0, 0);
        if constexpr (BTRANS)
          __builtin_prefetch((const void*)(Bg + (size_t)(n0 + (tid >> 2)) * ldb + (kb + 2) * BK), 0, 0);
        else
          __builtin_prefetch((const void*)(Bg + (size_t)((kb + 2) * BK + tid / (BN / 8)) * ldb + n0), 0, 0);
      }
    }

    // ---- gather fragments per ISA VGPR layout and issue WMMAs ----
    AB16 afr[MF], bfr[NF];
    for (int mf = 0; mf < MF; ++mf) {
      const unsigned short* rp = &lds_a[cur][(wm0 + mf * 16 + l16) * LSA];
      afr[mf].q[0] = *(const uint4*)(rp + half * 8);        // k = half*8 .. +8
      afr[mf].q[1] = *(const uint4*)(rp + 16 + half * 8);   // k = 16+half*8 .. +8
    }
    for (int nf = 0; nf < NF; ++nf) {
      const unsigned short* cp = &lds_b[cur][(wn0 + nf * 16 + l16) * LSB + half * 8];
      bfr[nf].q[0] = *(const uint4*)(cp);                   // k = half*8 .. +8
      bfr[nf].q[1] = *(const uint4*)(cp + 16);              // k = 16+half*8 .. +8
    }
    for (int mf = 0; mf < MF; ++mf)
      for (int nf = 0; nf < NF; ++nf)
        acc[mf][nf] = __builtin_amdgcn_wmma_f32_16x16x32_bf16(
            false, afr[mf].v, false, bfr[nf].v, (short)0, acc[mf][nf], false, false);

    if (more) lstore(cur ^ 1);                 // fill the other buffer
    __syncthreads();                           // single barrier per K-step
    cur ^= 1;
  }

  // ---- epilogue ----
  float dmax = 0.0f;
  for (int mf = 0; mf < MF; ++mf) {
    for (int nf = 0; nf < NF; ++nf) {
      const int gcol = n0 + wn0 + nf * 16 + l16;
      const int grow0 = m0 + wm0 + mf * 16 + half * 8;
      for (int r = 0; r < 8; ++r) {
        const int grow = grow0 + r;
        float v = acc[mf][nf][r];
        if constexpr (MODE == 0) {
          outbf[(size_t)grow * N + gcol] = f2bf(v + bias[gcol]);
        } else if constexpr (MODE == 1) {
          outf[(size_t)grow * N + gcol] = v + bias[gcol];
        } else if constexpr (MODE == 2) {
          size_t idx = (size_t)bh * (Ln * Ln) + (size_t)grow * Ln + gcol;
          float val = 0.98f * p0f[idx] + 0.02f * v;
          if (diffslot) dmax = fmaxf(dmax, fabsf(val - prevf[idx]));
          outf[idx] = val;
        } else if constexpr (MODE == 3) {
          outbf[(size_t)(bh >> 4) * (Ln * Dn) + (size_t)grow * Dn +
                (size_t)(bh & 15) * HDn + gcol] = f2bf(v);
        } else {  // MODE 4
          outf[(size_t)bh * (Ln * Ln) + (size_t)grow * Ln + gcol] = v * scale;
        }
      }
    }
  }
  if constexpr (MODE == 2) {
    if (diffslot) {
      for (int o = 16; o; o >>= 1) dmax = fmaxf(dmax, __shfl_xor(dmax, o, 32));
      if (lane == 0) atomicMax(diffslot, __float_as_uint(dmax));
    }
  }
}

// ---------------- host launcher ----------------

extern "C" void kernel_launch(void* const* d_in, const int* in_sizes, int n_in,
                              void* d_out, int out_size, void* d_ws, size_t ws_size,
                              hipStream_t stream) {
  (void)in_sizes; (void)n_in; (void)out_size; (void)ws_size;
  const float* hs = (const float*)d_in[0];
  const float* Wq = (const float*)d_in[1];  const float* bq = (const float*)d_in[2];
  const float* Wk = (const float*)d_in[3];  const float* bk = (const float*)d_in[4];
  const float* Wv = (const float*)d_in[5];  const float* bv = (const float*)d_in[6];
  const float* Wo = (const float*)d_in[7];  const float* bo = (const float*)d_in[8];
  float* out = (float*)d_out;

  char* w = (char*)d_ws;
  auto carve = [&](size_t bytes) -> char* {
    char* p = w;
    w += (bytes + 255) & ~(size_t)255;
    return p;
  };
  unsigned short* hsbf  = (unsigned short*)carve(NT * 2);
  unsigned short* Wqbf  = (unsigned short*)carve(WSZ * 2);
  unsigned short* Wkbf  = (unsigned short*)carve(WSZ * 2);
  unsigned short* Wvbf  = (unsigned short*)carve(WSZ * 2);
  unsigned short* Wobf  = (unsigned short*)carve(WSZ * 2);
  unsigned short* Qbf   = (unsigned short*)carve(NT * 2);
  unsigned short* Kbf   = (unsigned short*)carve(NT * 2);
  unsigned short* Vbf   = (unsigned short*)carve(NT * 2);
  unsigned short* Knbf  = (unsigned short*)carve(NT * 2);
  unsigned short* attnbf= (unsigned short*)carve(NT * 2);
  float*          p0f   = (float*)carve(PP * 4);
  float*          rawA  = (float*)carve(PP * 4);   // raw scores, then ping buffer pA
  float*          pBf   = (float*)carve(PP * 4);
  unsigned short* Pbf   = (unsigned short*)carve(PP * 2);
  unsigned int*   flags = (unsigned int*)carve(256);  // [0..3]=diffmax bits, [4]=done

  init_flags<<<1, 32, 0, stream>>>(flags);

  // fp32 -> bf16 conversions
  f2bf_kernel<<<(int)(NT / 256), 256, 0, stream>>>(hs, hsbf, (int)NT);
  f2bf_kernel<<<(int)(WSZ / 256), 256, 0, stream>>>(Wq, Wqbf, (int)WSZ);
  f2bf_kernel<<<(int)(WSZ / 256), 256, 0, stream>>>(Wk, Wkbf, (int)WSZ);
  f2bf_kernel<<<(int)(WSZ / 256), 256, 0, stream>>>(Wv, Wvbf, (int)WSZ);
  f2bf_kernel<<<(int)(WSZ / 256), 256, 0, stream>>>(Wo, Wobf, (int)WSZ);

  // Q/K/V projections: (4096x1024) @ (1024x1024) + bias -> bf16 (B,L,D)
  dim3 gproj(Dn / 128, MLD / 128, 1);
  gemm_kernel<128, 0, false, false><<<gproj, 256, 0, stream>>>(
      hsbf, Wqbf, bq, nullptr, nullptr, nullptr, Qbf, nullptr, nullptr,
      MLD, Dn, Dn, Dn, Dn, 1.0f);
  gemm_kernel<128, 0, false, false><<<gproj, 256, 0, stream>>>(
      hsbf, Wkbf, bk, nullptr, nullptr, nullptr, Kbf, nullptr, nullptr,
      MLD, Dn, Dn, Dn, Dn, 1.0f);
  gemm_kernel<128, 0, false, false><<<gproj, 256, 0, stream>>>(
      hsbf, Wvbf, bv, nullptr, nullptr, nullptr, Vbf, nullptr, nullptr,
      MLD, Dn, Dn, Dn, Dn, 1.0f);

  // Kn = K / max(||K||, eps)
  knorm_kernel<<<(Bn * Ln * Hn) / 8, 256, 0, stream>>>(Kbf, Knbf);

  dim3 gbatch(Ln / 128, Ln / 128, BHn);   // (4,4,128)
  // raw scores = Q @ K^T / 8
  gemm_kernel<128, 4, false, true><<<gbatch, 256, 0, stream>>>(
      Qbf, Kbf, nullptr, nullptr, nullptr, rawA, nullptr, nullptr, nullptr,
      Ln, Ln, HDn, Dn, Dn, 0.125f);
  softmax_kernel<<<(int)(BHn * Ln) / 8, 256, 0, stream>>>(rawA, p0f, nullptr, 0);

  // raw sim = Kn @ Kn^T / 8 ; P = jittered softmax (bf16 only)
  gemm_kernel<128, 4, false, true><<<gbatch, 256, 0, stream>>>(
      Knbf, Knbf, nullptr, nullptr, nullptr, rawA, nullptr, nullptr, nullptr,
      Ln, Ln, HDn, Dn, Dn, 0.125f);
  softmax_kernel<<<(int)(BHn * Ln) / 8, 256, 0, stream>>>(rawA, nullptr, Pbf, 1);

  // diffusion steps: p <- 0.98*p0 + 0.02*(p @ P), early stop via flags
  // step 0: p0 -> pA (no diff test)
  gemm_kernel<128, 2, true, false><<<gbatch, 256, 0, stream>>>(
      p0f, Pbf, nullptr, p0f, nullptr, rawA, nullptr, nullptr, flags + 4,
      Ln, Ln, Ln, Ln, Ln, 1.0f);
  // step 1: pA -> pB, diff vs p0
  gemm_kernel<128, 2, true, false><<<gbatch, 256, 0, stream>>>(
      rawA, Pbf, nullptr, p0f, p0f, pBf, nullptr, flags + 1, flags + 4,
      Ln, Ln, Ln, Ln, Ln, 1.0f);
  update_done<<<1, 1, 0, stream>>>(flags, 1);
  // step 2: pB -> pA, diff vs pA (old contents = p1)
  gemm_kernel<128, 2, true, false><<<gbatch, 256, 0, stream>>>(
      pBf, Pbf, nullptr, p0f, rawA, rawA, nullptr, flags + 2, flags + 4,
      Ln, Ln, Ln, Ln, Ln, 1.0f);
  update_done<<<1, 1, 0, stream>>>(flags, 2);
  // step 3: pA -> pB (no diff needed afterwards)
  gemm_kernel<128, 2, true, false><<<gbatch, 256, 0, stream>>>(
      rawA, Pbf, nullptr, p0f, nullptr, pBf, nullptr, nullptr, flags + 4,
      Ln, Ln, Ln, Ln, Ln, 1.0f);

  // attn = p @ V  -> bf16 in (B,L,D)
  dim3 gav(1, Ln / 128, BHn);
  gemm_kernel<64, 3, true, false><<<gav, 256, 0, stream>>>(
      pBf, Vbf, nullptr, nullptr, nullptr, nullptr, attnbf, nullptr, nullptr,
      Ln, HDn, Ln, Ln, Dn, 1.0f);

  // out = attn @ Wo + bo  -> fp32 d_out
  gemm_kernel<128, 1, false, false><<<gproj, 256, 0, stream>>>(
      attnbf, Wobf, bo, nullptr, nullptr, out, nullptr, nullptr, nullptr,
      MLD, Dn, Dn, Dn, Dn, 1.0f);
}
